// D_VAE_NET_9320079032866
// MI455X (gfx1250) — matrix operand order repaired and checked by reading, not compiled, not run on hardware
//
#include <hip/hip_runtime.h>
#include <hip/hip_bf16.h>
#include <stdint.h>

// ---------------------------------------------------------------------------
// D_VAE_NET for MI455X (gfx1250): bf16 WMMA GEMMs + f32 LSTM cell epilogues.
// Async global->LDS (ASYNCcnt) for activation tiles.
// ---------------------------------------------------------------------------

typedef __bf16 bf16_t;
typedef __attribute__((ext_vector_type(16))) __bf16 v16bf;
typedef __attribute__((ext_vector_type(8)))  float  v8f;
typedef __attribute__((ext_vector_type(4)))  int    v4i;

#define B_    256
#define T_    512
#define DIM_  256
#define ZDIM_ 128
#define ENC_  1024
#define DEC_  1024
#define KENC  (2 * DIM_ + ENC_)   // 1536: [x_t, x_hat, h_enc]
#define KDEC  (ZDIM_ + DEC_)      // 1152: [z, h_dec]

#if defined(__gfx1250__) && __has_builtin(__builtin_amdgcn_global_load_async_to_lds_b128)
#define HAVE_ASYNC_LDS 1
#else
#define HAVE_ASYNC_LDS 0
#endif

#define AS_GLOBAL __attribute__((address_space(1)))
#define AS_SHARED __attribute__((address_space(3)))

__device__ __forceinline__ float sigmoidf_(float x) {
    return 1.0f / (1.0f + __expf(-x));
}

__device__ __forceinline__ v8f wmma_bf16(v16bf a, v16bf b, v8f c) {
    // v_wmma_f32_16x16x32_bf16  D = A*B + C   (wave32)
    return __builtin_amdgcn_wmma_f32_16x16x32_bf16(
        /*neg_a=*/false, a, /*neg_b=*/false, b,
        /*c_mod=*/(short)0, c, /*reuse_a=*/false, /*reuse_b=*/false);
}

// 16-byte global -> LDS copy: async (ASYNCcnt, no VGPR data) when available.
__device__ __forceinline__ void copy16_g2l(const bf16_t* __restrict__ g,
                                           bf16_t* __restrict__ l) {
#if HAVE_ASYNC_LDS
    __builtin_amdgcn_global_load_async_to_lds_b128(
        (AS_GLOBAL v4i*)g, (AS_SHARED v4i*)l, 0, 0);
#else
    *(uint4*)l = *(const uint4*)g;
#endif
}

__device__ __forceinline__ void async_fill_fence() {
#if HAVE_ASYNC_LDS
    asm volatile("s_wait_asynccnt 0" ::: "memory");
#endif
}

// pack 4 f32 -> 4 bf16 (8 bytes)
union Pk4 { bf16_t h[4]; uint2 u; };

// ===========================================================================
// Encoder step: gates = [x_t, x_hat, h_enc] @ Wenc^T + bias, LSTM cell.
// grid (16 b-tiles, 16 n-groups), block 128 (4 waves, one n-tile each).
// ===========================================================================
__global__ __launch_bounds__(128) void k_enc_step(
    const float*  __restrict__ input,    // [B,T,DIM]
    const float*  __restrict__ cprev,    // [B,DIM]   (decoder output of t-1)
    const bf16_t* __restrict__ h_in,     // [B,ENC]   bank t&1
    const bf16_t* __restrict__ W,        // [4*ENC][KENC] bf16 row-major
    const float*  __restrict__ bias,     // [4*ENC]   b_ih + b_hh
    float*        __restrict__ c_state,  // [B,ENC]
    bf16_t*       __restrict__ h_out,    // [B,ENC]   bank (t+1)&1
    int t)
{
    __shared__ __align__(16) bf16_t As[16][KENC + 8];   // padded: conflict-free b128
    const int b0   = blockIdx.x * 16;
    const int wave = threadIdx.x >> 5;
    const int lane = threadIdx.x & 31;
    const int n0   = blockIdx.y * 64 + wave * 16;

    // --- x_t / x_hat regions: 16 rows x 128 quads of 4 floats
    for (int idx = threadIdx.x; idx < 16 * 128; idx += 128) {
        const int r = idx >> 7;
        const int q = idx & 127;
        const int b = b0 + r;
        if (q < 64) {                         // x_t: cols 4q..4q+3
            const int k = 4 * q;
            const float4 x = *(const float4*)&input[((size_t)b * T_ + t) * DIM_ + k];
            Pk4 p; p.h[0] = (bf16_t)x.x; p.h[1] = (bf16_t)x.y;
                   p.h[2] = (bf16_t)x.z; p.h[3] = (bf16_t)x.w;
            *(uint2*)&As[r][k] = p.u;
        } else {                              // x_hat = x - sigmoid(cprev)
            const int d = 4 * (q - 64);
            const float4 x = *(const float4*)&input[((size_t)b * T_ + t) * DIM_ + d];
            const float4 c = *(const float4*)&cprev[b * DIM_ + d];
            Pk4 p;
            p.h[0] = (bf16_t)(x.x - sigmoidf_(c.x));
            p.h[1] = (bf16_t)(x.y - sigmoidf_(c.y));
            p.h[2] = (bf16_t)(x.z - sigmoidf_(c.z));
            p.h[3] = (bf16_t)(x.w - sigmoidf_(c.w));
            *(uint2*)&As[r][DIM_ + d] = p.u;
        }
    }
    // --- h_enc region: pure copy, 16 rows x 128 x (8 bf16) via async-to-LDS
    for (int idx = threadIdx.x; idx < 16 * (ENC_ / 8); idx += 128) {
        const int r = idx >> 7;
        const int k = 8 * (idx & 127);
        copy16_g2l(&h_in[(size_t)(b0 + r) * ENC_ + k], &As[r][2 * DIM_ + k]);
    }
    async_fill_fence();
    __syncthreads();

    const int m  = lane & 15;
    const int hi = lane >> 4;

    v8f acc0 = {0,0,0,0,0,0,0,0};
    v8f acc1 = acc0, acc2 = acc0, acc3 = acc0;

    const bf16_t* Arow = &As[m][0];
    const bf16_t* B0 = W + (size_t)(0 * ENC_ + n0 + m) * KENC + hi * 16;
    const bf16_t* B1 = W + (size_t)(1 * ENC_ + n0 + m) * KENC + hi * 16;
    const bf16_t* B2 = W + (size_t)(2 * ENC_ + n0 + m) * KENC + hi * 16;
    const bf16_t* B3 = W + (size_t)(3 * ENC_ + n0 + m) * KENC + hi * 16;

    for (int k0 = 0; k0 < KENC; k0 += 32) {
        union { v16bf v; uint4 q[2]; } a, q0, q1, q2, q3;
        a.q[0] = *(const uint4*)&Arow[k0 + hi * 16];          // ds_load_b128
        a.q[1] = *(const uint4*)&Arow[k0 + hi * 16 + 8];
        q0.q[0] = *(const uint4*)&B0[k0]; q0.q[1] = *(const uint4*)&B0[k0 + 8];
        q1.q[0] = *(const uint4*)&B1[k0]; q1.q[1] = *(const uint4*)&B1[k0 + 8];
        q2.q[0] = *(const uint4*)&B2[k0]; q2.q[1] = *(const uint4*)&B2[k0 + 8];
        q3.q[0] = *(const uint4*)&B3[k0]; q3.q[1] = *(const uint4*)&B3[k0 + 8];
        acc0 = wmma_bf16(a.v, q0.v, acc0);   // i gate
        acc1 = wmma_bf16(a.v, q1.v, acc1);   // f gate
        acc2 = wmma_bf16(a.v, q2.v, acc2);   // g gate
        acc3 = wmma_bf16(a.v, q3.v, acc3);   // o gate
    }

    const int n  = n0 + m;
    const float bi = bias[0 * ENC_ + n];
    const float bf = bias[1 * ENC_ + n];
    const float bg = bias[2 * ENC_ + n];
    const float bo = bias[3 * ENC_ + n];
#pragma unroll
    for (int r = 0; r < 8; ++r) {
        const int    b  = b0 + r + hi * 8;    // C/D layout: lanes16-31 -> M=8+r
        const size_t ix = (size_t)b * ENC_ + n;
        const float iv = sigmoidf_(acc0[r] + bi);
        const float fv = sigmoidf_(acc1[r] + bf);
        const float gv = tanhf(acc2[r] + bg);
        const float ov = sigmoidf_(acc3[r] + bo);
        const float cn = fv * c_state[ix] + iv * gv;
        c_state[ix] = cn;
        h_out[ix]   = (bf16_t)(ov * tanhf(cn));
    }
}

// ===========================================================================
// Latent: mu/logsigma = h_enc @ {w_mu,w_sigma} + b; z = mu + exp(ls)*e.
// grid (16, 2), block 128.
// ===========================================================================
__global__ __launch_bounds__(128) void k_latent(
    const bf16_t* __restrict__ h_in,   // [B,ENC]
    const bf16_t* __restrict__ Wmu,    // [ZDIM][ENC] (pre-transposed)
    const bf16_t* __restrict__ Wsg,    // [ZDIM][ENC]
    const float*  __restrict__ bmu,
    const float*  __restrict__ bsg,
    const float*  __restrict__ e,      // [B,ZDIM]
    float* __restrict__ out_sigma, float* __restrict__ out_mu,
    float* __restrict__ out_ls,    float* __restrict__ out_z,
    bf16_t* __restrict__ z_bf, int t)
{
    __shared__ __align__(16) bf16_t As[16][ENC_ + 8];
    const int b0   = blockIdx.x * 16;
    const int wave = threadIdx.x >> 5;
    const int lane = threadIdx.x & 31;
    const int n0   = blockIdx.y * 64 + wave * 16;

    for (int idx = threadIdx.x; idx < 16 * (ENC_ / 8); idx += 128) {
        const int r = idx >> 7;
        const int k = 8 * (idx & 127);
        copy16_g2l(&h_in[(size_t)(b0 + r) * ENC_ + k], &As[r][k]);
    }
    async_fill_fence();
    __syncthreads();

    const int m = lane & 15, hi = lane >> 4;
    v8f am = {0,0,0,0,0,0,0,0};
    v8f al = am;

    const bf16_t* Arow = &As[m][0];
    const bf16_t* Bm = Wmu + (size_t)(n0 + m) * ENC_ + hi * 16;
    const bf16_t* Bs = Wsg + (size_t)(n0 + m) * ENC_ + hi * 16;

    for (int k0 = 0; k0 < ENC_; k0 += 32) {
        union { v16bf v; uint4 q[2]; } a, qm, qs;
        a.q[0] = *(const uint4*)&Arow[k0 + hi * 16];
        a.q[1] = *(const uint4*)&Arow[k0 + hi * 16 + 8];
        qm.q[0] = *(const uint4*)&Bm[k0]; qm.q[1] = *(const uint4*)&Bm[k0 + 8];
        qs.q[0] = *(const uint4*)&Bs[k0]; qs.q[1] = *(const uint4*)&Bs[k0 + 8];
        am = wmma_bf16(a.v, qm.v, am);
        al = wmma_bf16(a.v, qs.v, al);
    }

    const int n = n0 + m;
    const float bmv = bmu[n], bsv = bsg[n];
#pragma unroll
    for (int r = 0; r < 8; ++r) {
        const int b = b0 + r + hi * 8;
        const float mu = am[r] + bmv;
        const float ls = al[r] + bsv;
        const float sg = __expf(ls);
        const float z  = mu + sg * e[b * ZDIM_ + n];
        const size_t o = ((size_t)b * T_ + t) * ZDIM_ + n;
        out_sigma[o] = sg; out_mu[o] = mu; out_ls[o] = ls; out_z[o] = z;
        z_bf[b * ZDIM_ + n] = (bf16_t)z;
    }
}

// ===========================================================================
// Decoder step: gates = [z, h_dec] @ Wdec^T + bias, LSTM cell.
// grid (16, 16), block 128.
// ===========================================================================
__global__ __launch_bounds__(128) void k_dec_step(
    const bf16_t* __restrict__ z_bf,   // [B,ZDIM]
    const bf16_t* __restrict__ h_in,   // [B,DEC] bank t&1
    const bf16_t* __restrict__ W,      // [4*DEC][KDEC]
    const float*  __restrict__ bias,   // [4*DEC]
    float*        __restrict__ c_state,
    bf16_t*       __restrict__ h_out)  // bank (t+1)&1
{
    __shared__ __align__(16) bf16_t As[16][KDEC + 8];
    const int b0   = blockIdx.x * 16;
    const int wave = threadIdx.x >> 5;
    const int lane = threadIdx.x & 31;
    const int n0   = blockIdx.y * 64 + wave * 16;

    // pure copy fill: [z | h_dec], 16 rows x 144 x (8 bf16)
    for (int idx = threadIdx.x; idx < 16 * (KDEC / 8); idx += 128) {
        const int r = idx / (KDEC / 8);
        const int k = 8 * (idx - r * (KDEC / 8));
        const int b = b0 + r;
        const bf16_t* src = (k < ZDIM_)
            ? &z_bf[(size_t)b * ZDIM_ + k]
            : &h_in[(size_t)b * DEC_ + (k - ZDIM_)];
        copy16_g2l(src, &As[r][k]);
    }
    async_fill_fence();
    __syncthreads();

    const int m = lane & 15, hi = lane >> 4;
    v8f acc0 = {0,0,0,0,0,0,0,0};
    v8f acc1 = acc0, acc2 = acc0, acc3 = acc0;

    const bf16_t* Arow = &As[m][0];
    const bf16_t* B0 = W + (size_t)(0 * DEC_ + n0 + m) * KDEC + hi * 16;
    const bf16_t* B1 = W + (size_t)(1 * DEC_ + n0 + m) * KDEC + hi * 16;
    const bf16_t* B2 = W + (size_t)(2 * DEC_ + n0 + m) * KDEC + hi * 16;
    const bf16_t* B3 = W + (size_t)(3 * DEC_ + n0 + m) * KDEC + hi * 16;

    for (int k0 = 0; k0 < KDEC; k0 += 32) {
        union { v16bf v; uint4 q[2]; } a, q0, q1, q2, q3;
        a.q[0] = *(const uint4*)&Arow[k0 + hi * 16];
        a.q[1] = *(const uint4*)&Arow[k0 + hi * 16 + 8];
        q0.q[0] = *(const uint4*)&B0[k0]; q0.q[1] = *(const uint4*)&B0[k0 + 8];
        q1.q[0] = *(const uint4*)&B1[k0]; q1.q[1] = *(const uint4*)&B1[k0 + 8];
        q2.q[0] = *(const uint4*)&B2[k0]; q2.q[1] = *(const uint4*)&B2[k0 + 8];
        q3.q[0] = *(const uint4*)&B3[k0]; q3.q[1] = *(const uint4*)&B3[k0 + 8];
        acc0 = wmma_bf16(a.v, q0.v, acc0);
        acc1 = wmma_bf16(a.v, q1.v, acc1);
        acc2 = wmma_bf16(a.v, q2.v, acc2);
        acc3 = wmma_bf16(a.v, q3.v, acc3);
    }

    const int n = n0 + m;
    const float bi = bias[0 * DEC_ + n];
    const float bf = bias[1 * DEC_ + n];
    const float bg = bias[2 * DEC_ + n];
    const float bo = bias[3 * DEC_ + n];
#pragma unroll
    for (int r = 0; r < 8; ++r) {
        const int    b  = b0 + r + hi * 8;
        const size_t ix = (size_t)b * DEC_ + n;
        const float iv = sigmoidf_(acc0[r] + bi);
        const float fv = sigmoidf_(acc1[r] + bf);
        const float gv = tanhf(acc2[r] + bg);
        const float ov = sigmoidf_(acc3[r] + bo);
        const float cn = fv * c_state[ix] + iv * gv;
        c_state[ix] = cn;
        h_out[ix]   = (bf16_t)(ov * tanhf(cn));
    }
}

// ===========================================================================
// Output projection: c_out = sigmoid(h_dec @ w_h_dec + b).  grid (16, 4).
// ===========================================================================
__global__ __launch_bounds__(128) void k_outproj(
    const bf16_t* __restrict__ h_in,   // [B,DEC]
    const bf16_t* __restrict__ Wt,     // [DIM][DEC] (pre-transposed)
    const float*  __restrict__ bias,   // [DIM]
    float* __restrict__ out_dec,       // [B,T,DIM]
    float* __restrict__ cprev,         // [B,DIM]
    int t)
{
    __shared__ __align__(16) bf16_t As[16][DEC_ + 8];
    const int b0   = blockIdx.x * 16;
    const int wave = threadIdx.x >> 5;
    const int lane = threadIdx.x & 31;
    const int n0   = blockIdx.y * 64 + wave * 16;

    for (int idx = threadIdx.x; idx < 16 * (DEC_ / 8); idx += 128) {
        const int r = idx >> 7;
        const int k = 8 * (idx & 127);
        copy16_g2l(&h_in[(size_t)(b0 + r) * DEC_ + k], &As[r][k]);
    }
    async_fill_fence();
    __syncthreads();

    const int m = lane & 15, hi = lane >> 4;
    v8f acc = {0,0,0,0,0,0,0,0};
    const bf16_t* Arow = &As[m][0];
    const bf16_t* Bp = Wt + (size_t)(n0 + m) * DEC_ + hi * 16;

    for (int k0 = 0; k0 < DEC_; k0 += 32) {
        union { v16bf v; uint4 q[2]; } a, qb;
        a.q[0] = *(const uint4*)&Arow[k0 + hi * 16];
        a.q[1] = *(const uint4*)&Arow[k0 + hi * 16 + 8];
        qb.q[0] = *(const uint4*)&Bp[k0]; qb.q[1] = *(const uint4*)&Bp[k0 + 8];
        acc = wmma_bf16(a.v, qb.v, acc);
    }

    const int n = n0 + m;
    const float bv = bias[n];
#pragma unroll
    for (int r = 0; r < 8; ++r) {
        const int b = b0 + r + hi * 8;
        const float s = sigmoidf_(acc[r] + bv);
        out_dec[((size_t)b * T_ + t) * DIM_ + n] = s;
        cprev[b * DIM_ + n] = s;
    }
}

// ============================ init / pack kernels ==========================

__global__ void k_zero32(float* __restrict__ p, int n) {
    int i = blockIdx.x * blockDim.x + threadIdx.x;
    if (i < n) p[i] = 0.0f;
}

// W[r][0:Ka] = A[r][:], W[r][Ka:Ka+Kb] = Bm[r][:]  (f32 -> bf16)
__global__ void k_pack2(const float* __restrict__ A, const float* __restrict__ Bm,
                        bf16_t* __restrict__ W, int R, int Ka, int Kb) {
    const int K = Ka + Kb;
    int i = blockIdx.x * blockDim.x + threadIdx.x;
    if (i >= R * K) return;
    const int r = i / K, k = i - r * K;
    const float v = (k < Ka) ? A[(size_t)r * Ka + k] : Bm[(size_t)r * Kb + (k - Ka)];
    W[i] = (bf16_t)v;
}

// A[K][N] (f32) -> W[N][K] (bf16)
__global__ void k_transpose(const float* __restrict__ A, bf16_t* __restrict__ W,
                            int K, int N) {
    int i = blockIdx.x * blockDim.x + threadIdx.x;
    if (i >= K * N) return;
    const int n = i / K, k = i - n * K;
    W[i] = (bf16_t)A[(size_t)k * N + n];
}

__global__ void k_addvec(const float* __restrict__ a, const float* __restrict__ b,
                         float* __restrict__ o, int n) {
    int i = blockIdx.x * blockDim.x + threadIdx.x;
    if (i < n) o[i] = a[i] + b[i];
}

// ================================ launcher =================================

extern "C" void kernel_launch(void* const* d_in, const int* in_sizes, int n_in,
                              void* d_out, int out_size, void* d_ws, size_t ws_size,
                              hipStream_t stream) {
    const float* input    = (const float*)d_in[0];
    const float* W_ih_enc = (const float*)d_in[1];
    const float* W_hh_enc = (const float*)d_in[2];
    const float* b_ih_enc = (const float*)d_in[3];
    const float* b_hh_enc = (const float*)d_in[4];
    const float* W_ih_dec = (const float*)d_in[5];
    const float* W_hh_dec = (const float*)d_in[6];
    const float* b_ih_dec = (const float*)d_in[7];
    const float* b_hh_dec = (const float*)d_in[8];
    const float* w_mu     = (const float*)d_in[9];
    const float* b_mu     = (const float*)d_in[10];
    const float* w_sigma  = (const float*)d_in[11];
    const float* b_sigma  = (const float*)d_in[12];
    const float* w_h_dec  = (const float*)d_in[13];
    const float* b_h_dec  = (const float*)d_in[14];
    const float* e        = (const float*)d_in[15];

    // ---- workspace carve-up (all sizes multiples of 256B; state is contiguous)
    char* ws = (char*)d_ws;
    size_t off = 0;
    auto alloc = [&](size_t bytes) -> void* {
        void* p = ws + off;
        off += (bytes + 255) & ~(size_t)255;
        return p;
    };
    float*  c_enc  = (float*)alloc((size_t)B_ * ENC_ * 4);
    float*  c_dec  = (float*)alloc((size_t)B_ * DEC_ * 4);
    float*  cprev  = (float*)alloc((size_t)B_ * DIM_ * 4);
    bf16_t* h_enc_b[2] = { (bf16_t*)alloc((size_t)B_ * ENC_ * 2),
                           (bf16_t*)alloc((size_t)B_ * ENC_ * 2) };
    bf16_t* h_dec_b[2] = { (bf16_t*)alloc((size_t)B_ * DEC_ * 2),
                           (bf16_t*)alloc((size_t)B_ * DEC_ * 2) };
    const int zero_n = (int)(off / 4);       // zero c_enc .. h_dec banks span
    bf16_t* z_bf  = (bf16_t*)alloc((size_t)B_ * ZDIM_ * 2);
    bf16_t* Wenc  = (bf16_t*)alloc((size_t)4 * ENC_ * KENC * 2);
    bf16_t* Wdec  = (bf16_t*)alloc((size_t)4 * DEC_ * KDEC * 2);
    bf16_t* Wmu   = (bf16_t*)alloc((size_t)ZDIM_ * ENC_ * 2);
    bf16_t* Wsg   = (bf16_t*)alloc((size_t)ZDIM_ * ENC_ * 2);
    bf16_t* Whd   = (bf16_t*)alloc((size_t)DIM_ * DEC_ * 2);
    float*  benc  = (float*)alloc((size_t)4 * ENC_ * 4);
    float*  bdec  = (float*)alloc((size_t)4 * DEC_ * 4);

    // ---- output slices (decoded, sigma, mu, logsigma, z) — flat concat
    float* out       = (float*)d_out;
    float* out_dec   = out;
    float* out_sigma = out_dec   + (size_t)B_ * T_ * DIM_;
    float* out_mu    = out_sigma + (size_t)B_ * T_ * ZDIM_;
    float* out_ls    = out_mu    + (size_t)B_ * T_ * ZDIM_;
    float* out_z     = out_ls    + (size_t)B_ * T_ * ZDIM_;

    // ---- init: zero recurrent state, pack weights to bf16
    k_zero32<<<(zero_n + 255) / 256, 256, 0, stream>>>(c_enc, zero_n);
    {
        int n = 4 * ENC_ * KENC;
        k_pack2<<<(n + 255) / 256, 256, 0, stream>>>(W_ih_enc, W_hh_enc, Wenc,
                                                     4 * ENC_, 2 * DIM_, ENC_);
        n = 4 * DEC_ * KDEC;
        k_pack2<<<(n + 255) / 256, 256, 0, stream>>>(W_ih_dec, W_hh_dec, Wdec,
                                                     4 * DEC_, ZDIM_, DEC_);
        n = ZDIM_ * ENC_;
        k_transpose<<<(n + 255) / 256, 256, 0, stream>>>(w_mu, Wmu, ENC_, ZDIM_);
        k_transpose<<<(n + 255) / 256, 256, 0, stream>>>(w_sigma, Wsg, ENC_, ZDIM_);
        n = DIM_ * DEC_;
        k_transpose<<<(n + 255) / 256, 256, 0, stream>>>(w_h_dec, Whd, DEC_, DIM_);
        n = 4 * ENC_;
        k_addvec<<<(n + 255) / 256, 256, 0, stream>>>(b_ih_enc, b_hh_enc, benc, n);
        n = 4 * DEC_;
        k_addvec<<<(n + 255) / 256, 256, 0, stream>>>(b_ih_dec, b_hh_dec, bdec, n);
    }

    // ---- sequential recurrence: 4 WMMA kernels per time step
    for (int t = 0; t < T_; ++t) {
        const int p = t & 1;
        k_enc_step<<<dim3(16, 16), 128, 0, stream>>>(
            input, cprev, h_enc_b[p], Wenc, benc, c_enc, h_enc_b[1 - p], t);
        k_latent<<<dim3(16, 2), 128, 0, stream>>>(
            h_enc_b[1 - p], Wmu, Wsg, b_mu, b_sigma, e,
            out_sigma, out_mu, out_ls, out_z, z_bf, t);
        k_dec_step<<<dim3(16, 16), 128, 0, stream>>>(
            z_bf, h_dec_b[p], Wdec, bdec, c_dec, h_dec_b[1 - p]);
        k_outproj<<<dim3(16, 4), 128, 0, stream>>>(
            h_dec_b[1 - p], Whd, b_h_dec, out_dec, cprev, t);
    }
    (void)in_sizes; (void)n_in; (void)out_size; (void)ws_size;
}
